// DynamicNet_66451734003712
// MI455X (gfx1250) — matrix operand order repaired and checked by reading, not compiled, not run on hardware
//
#include <hip/hip_runtime.h>
#include <math.h>
#include <stdint.h>

typedef __bf16 bf16_t;
typedef __attribute__((ext_vector_type(16))) __bf16 v16bf;
typedef __attribute__((ext_vector_type(8)))  float  v8f;
typedef __attribute__((ext_vector_type(4)))  unsigned int u32x4;
typedef __attribute__((ext_vector_type(8)))  int         i32x8;
typedef __attribute__((ext_vector_type(4)))  int         i32x4;

#define K_T   16
#define NE    2048
#define NU    4096
#define BATCH 8192

#if defined(__gfx1250__) && __has_builtin(__builtin_amdgcn_tensor_load_to_lds) && __has_builtin(__builtin_amdgcn_s_wait_tensorcnt)
#define USE_TDM 1
#else
#define USE_TDM 0
#endif

__device__ __forceinline__ float sigf(float x){ return 1.0f/(1.0f+__expf(-x)); }

#if USE_TDM
// ---- Tensor Data Mover: 2D bf16 tile (tile_d0 elems/row x tile_d1 rows),
// global row pitch = stride_elems, LDS rows padded +4 dwords every 16 dwords.
__device__ __forceinline__ void tdm_load_tile_2d(
    unsigned int lds_off, const void* gptr, unsigned int stride_elems,
    unsigned int tile_d0, unsigned int tile_d1)
{
    unsigned long long ga = (unsigned long long)(uintptr_t)gptr;
    u32x4 g0;
    g0[0] = 1u;                                   // count=1 (valid user D#)
    g0[1] = lds_off;                              // lds_addr (bytes)
    g0[2] = (unsigned int)(ga & 0xffffffffu);     // global_addr[31:0]
    g0[3] = (unsigned int)((ga >> 32) & 0x01ffffffu) | (2u << 30); // addr[56:32] | type=2
    i32x8 g1;
    g1[0] = (int)((1u << 16)          // data_size = 1 -> 2 bytes
                | (1u << 20)          // pad_enable
                | (3u << 22)          // pad_interval code 3 -> every 16 dwords
                | (3u << 25));        // pad_amount  code 3 -> 4 dwords
    g1[1] = (int)0xffff0000u;                     // abar_addr=0 | tensor_dim0 lo16
    g1[2] = (int)0xffff0fffu;                     // tensor_dim0 hi16 | tensor_dim1 lo16
    g1[3] = (int)((tile_d0 << 16) | 0x0fffu);     // tensor_dim1 hi16 | tile_dim0
    g1[4] = (int)(tile_d1 & 0xffffu);             // tile_dim1 | tile_dim2=0
    g1[5] = (int)stride_elems;                    // tensor_dim0_stride[31:0]
    g1[6] = 0;                                    // stride hi | dim1_stride lo
    g1[7] = 0;
    i32x4 z4 = {0,0,0,0};
#if __has_include(<hip/amd_detail/amd_gfx1250_TDM.h>)
    i32x8 z8 = {0,0,0,0,0,0,0,0};
    __builtin_amdgcn_tensor_load_to_lds(g0, g1, z4, z4, z8, 0);
#else
    __builtin_amdgcn_tensor_load_to_lds(g0, g1, z4, z4, 0);
#endif
}
#endif

// ---------------------------------------------------------------- LSTM
__global__ void __launch_bounds__(256) lstm_kernel(
    const float* __restrict__ tra, const float* __restrict__ usr_tab,
    const float* __restrict__ Wih, const float* __restrict__ Whh,
    const float* __restrict__ bih, const float* __restrict__ bhh,
    float* __restrict__ traF)
{
    __shared__ float sWih[64*12];
    __shared__ float sWhh[64*16];
    __shared__ float sB[64];
    int tid = threadIdx.x;
    for (int i = tid; i < 64*12; i += 256) sWih[i] = Wih[i];
    for (int i = tid; i < 64*16; i += 256) sWhh[i] = Whh[i];
    if (tid < 64) sB[tid] = bih[tid] + bhh[tid];
    __syncthreads();
    int u = blockIdx.x*256 + tid;

    float xt[12];
    #pragma unroll
    for (int j = 0; j < 8; j++) xt[j] = usr_tab[u*8 + j];
    float h[16], c[16];
    #pragma unroll
    for (int q = 0; q < 16; q++) { h[q] = 0.f; c[q] = 0.f; }

    for (int t = 0; t < K_T; t++) {
        #pragma unroll
        for (int j = 0; j < 4; j++) xt[8+j] = tra[((size_t)t*NU + u)*4 + j];
        float hn[16], cn[16];
        #pragma unroll
        for (int q = 0; q < 16; q++) {
            float zi = sB[q], zf = sB[16+q], zg = sB[32+q], zo = sB[48+q];
            #pragma unroll
            for (int cc = 0; cc < 12; cc++) {
                float xv = xt[cc];
                zi += sWih[q*12+cc]*xv;      zf += sWih[(16+q)*12+cc]*xv;
                zg += sWih[(32+q)*12+cc]*xv; zo += sWih[(48+q)*12+cc]*xv;
            }
            #pragma unroll
            for (int cc = 0; cc < 16; cc++) {
                float hv = h[cc];
                zi += sWhh[q*16+cc]*hv;      zf += sWhh[(16+q)*16+cc]*hv;
                zg += sWhh[(32+q)*16+cc]*hv; zo += sWhh[(48+q)*16+cc]*hv;
            }
            cn[q] = sigf(zf)*c[q] + sigf(zi)*tanhf(zg);
            hn[q] = sigf(zo)*tanhf(cn[q]);
        }
        #pragma unroll
        for (int q = 0; q < 16; q++) { h[q] = hn[q]; c[q] = cn[q]; }
    }
    #pragma unroll
    for (int q = 0; q < 16; q++) traF[(size_t)u*16 + q] = h[q];
}

// ---------------------------------------------------------------- edge projections
__global__ void __launch_bounds__(256) ep_kernel(
    const float* __restrict__ edge,
    const float* __restrict__ Wp, const float* __restrict__ bp,
    const float* __restrict__ Wa, const float* __restrict__ ba,
    float* __restrict__ epP, float* __restrict__ epA)
{
    int g = blockIdx.x*256 + threadIdx.x;
    float ev[11];
    #pragma unroll
    for (int cc = 0; cc < 11; cc++) ev[cc] = edge[(size_t)g*11 + cc];
    #pragma unroll
    for (int o = 0; o < 8; o++) {
        float ap = bp[o], aa = ba[o];
        #pragma unroll
        for (int cc = 0; cc < 11; cc++) { ap += Wp[o*11+cc]*ev[cc]; aa += Wa[o*11+cc]*ev[cc]; }
        epP[(size_t)g*8 + o] = ap;
        epA[(size_t)g*8 + o] = aa;
    }
}

// ---------------------------------------------------------------- T = ep @ (E2[t]·E1)
__global__ void __launch_bounds__(256) tmat_kernel(
    const float* __restrict__ ep, const float* __restrict__ E1,
    const float* __restrict__ E2, float* __restrict__ T)
{
    __shared__ float sM[64];
    int tid = threadIdx.x;
    int t = blockIdx.y;
    if (tid < 64) {
        int u = tid >> 3, v = tid & 7;
        float acc = 0.f;
        #pragma unroll
        for (int o = 0; o < 8; o++) acc += E2[t*8+o] * E1[(o*8+u)*8+v];
        sM[tid] = acc;
    }
    __syncthreads();
    int i = blockIdx.x*256 + tid;
    float ev[8];
    #pragma unroll
    for (int v = 0; v < 8; v++) ev[v] = ep[((size_t)t*NE + i)*8 + v];
    #pragma unroll
    for (int v = 0; v < 8; v++) {
        float acc = 0.f;
        #pragma unroll
        for (int u = 0; u < 8; u++) acc += ev[u]*sM[u*8+v];
        T[((size_t)t*NE + i)*8 + v] = acc;
    }
}

// ---------------------------------------------------------------- 3->32 projections
__global__ void __launch_bounds__(256) proj_kernel(
    const float* __restrict__ loadx, const float* __restrict__ svcx,
    const float* __restrict__ Wp, const float* __restrict__ bp,
    const float* __restrict__ Wa, const float* __restrict__ ba,
    float* __restrict__ xp, float* __restrict__ xa)
{
    int g = blockIdx.x*256 + threadIdx.x;
    float lp[3], la[3];
    #pragma unroll
    for (int cc = 0; cc < 3; cc++) { lp[cc] = loadx[(size_t)g*3+cc]; la[cc] = svcx[(size_t)g*3+cc]; }
    #pragma unroll
    for (int f = 0; f < 32; f++) {
        float ap = bp[f], aa = ba[f];
        #pragma unroll
        for (int cc = 0; cc < 3; cc++) { ap += Wp[f*3+cc]*lp[cc]; aa += Wa[f*3+cc]*la[cc]; }
        xp[(size_t)g*32+f] = ap;
        xa[(size_t)g*32+f] = aa;
    }
}

// ---------------------------------------------------------------- S row + relu + softmax -> bf16 A row
__global__ void __launch_bounds__(256) adj_kernel(
    const float* __restrict__ T, const float* __restrict__ ep,
    bf16_t* __restrict__ Adj)
{
    __shared__ float sS[NE];
    __shared__ float red[256];
    int tid = threadIdx.x;
    int i = blockIdx.x, t = blockIdx.y;
    float Tv[8];
    #pragma unroll
    for (int v = 0; v < 8; v++) Tv[v] = T[((size_t)t*NE + i)*8 + v];
    const float* epb = ep + (size_t)t*NE*8;

    float mx = 0.0f;
    for (int j = tid; j < NE; j += 256) {
        const float* e = epb + (size_t)j*8;
        float d = 0.f;
        #pragma unroll
        for (int v = 0; v < 8; v++) d += Tv[v]*e[v];
        d = fmaxf(d, 0.f);
        sS[j] = d;
        mx = fmaxf(mx, d);
    }
    red[tid] = mx; __syncthreads();
    for (int s = 128; s > 0; s >>= 1) { if (tid < s) red[tid] = fmaxf(red[tid], red[tid+s]); __syncthreads(); }
    mx = red[0];
    __syncthreads();

    float sum = 0.f;
    for (int j = tid; j < NE; j += 256) {
        float e = __expf(sS[j] - mx);
        sS[j] = e;
        sum += e;
    }
    red[tid] = sum; __syncthreads();
    for (int s = 128; s > 0; s >>= 1) { if (tid < s) red[tid] += red[tid+s]; __syncthreads(); }
    float inv = 1.0f / red[0];
    bf16_t* out = Adj + ((size_t)t*NE + i)*NE;
    for (int j = tid; j < NE; j += 256) out[j] = (bf16_t)(sS[j]*inv);
}

// ---------------------------------------------------------------- 64x64 LDS tile transpose: AdjT[t][m][n] = Adj[t][n][m]
__global__ void __launch_bounds__(256) transpose_kernel(
    const bf16_t* __restrict__ A, bf16_t* __restrict__ AT)
{
    __shared__ bf16_t tile[64][66];
    int tid = threadIdx.x;
    int t  = blockIdx.z;
    int i0 = blockIdx.x * 64;     // source row band
    int j0 = blockIdx.y * 64;     // source col band
    const unsigned int* src = (const unsigned int*)A;
    unsigned int* sD = (unsigned int*)&tile[0][0];     // row pitch 33 dwords
    #pragma unroll
    for (int q = 0; q < 8; q++) {
        int u = tid*8 + q;
        int r = u >> 5, cu = u & 31;                   // 32 dwords per 64-elem row
        sD[r*33 + cu] = src[(((size_t)t*NE + i0 + r)*NE + j0)/2 + cu];
    }
    __syncthreads();
    unsigned int* dst = (unsigned int*)AT;
    #pragma unroll
    for (int q = 0; q < 8; q++) {
        int u = tid*8 + q;
        int r = u >> 5, cu = u & 31;
        union { bf16_t h[2]; unsigned int u32; } pk;
        pk.h[0] = tile[2*cu][r];
        pk.h[1] = tile[2*cu+1][r];
        dst[(((size_t)t*NE + j0 + r)*NE + i0)/2 + cu] = pk.u32;
    }
}

// ---------------------------------------------------------------- dilated temporal conv -> relu -> yT (bf16, transposed)
__global__ void __launch_bounds__(256) conv_kernel(
    const float* __restrict__ xin, const float* __restrict__ Wt,
    const float* __restrict__ bt, int dil, bf16_t* __restrict__ yT)
{
    __shared__ float sW[3*32*32];
    __shared__ float sB[32];
    int tid = threadIdx.x;
    for (int i = tid; i < 3072; i += 256) sW[i] = Wt[i];
    if (tid < 32) sB[tid] = bt[tid];
    __syncthreads();
    int g = blockIdx.x*256 + tid;
    int t = g >> 11, n = g & (NE-1);

    float xr[3][32];
    #pragma unroll
    for (int j = 0; j < 3; j++) {
        int ts = t - (2-j)*dil;
        if (ts >= 0) {
            const float* p = xin + ((size_t)ts*NE + n)*32;
            #pragma unroll
            for (int f = 0; f < 32; f++) xr[j][f] = p[f];
        } else {
            #pragma unroll
            for (int f = 0; f < 32; f++) xr[j][f] = 0.f;
        }
    }
    for (int gg = 0; gg < 32; gg++) {
        float acc = sB[gg];
        #pragma unroll
        for (int j = 0; j < 3; j++)
            #pragma unroll
            for (int f = 0; f < 32; f++) acc += xr[j][f]*sW[(j*32+f)*32+gg];
        acc = fmaxf(acc, 0.f);
        yT[((size_t)t*32 + gg)*NE + n] = (bf16_t)acc;
    }
}

// ---------------------------------------------------------------- graph mix (WMMA bf16, double-buffered TDM pipeline)
// D[m][f] = sum_n AdjT[t][m][n] * y[t][n][f];  x_next[m][g] = relu(sum_f D[m][f]*Wg[f][g] + bg[g])
__global__ void __launch_bounds__(256) mix_kernel(
    const bf16_t* __restrict__ AdjT, const bf16_t* __restrict__ yT,
    const float* __restrict__ Wg, const float* __restrict__ bg,
    float* __restrict__ xout)
{
    __shared__ bf16_t sA[2][128][40];  // double-buffered 128 m-rows x 32 n, pitch 80B (TDM pad)
    __shared__ bf16_t sY[2][32][40];   // double-buffered 32 f-rows x 32 n
    __shared__ float  sMix[128][33];
    __shared__ float  sWg[32*32];
    __shared__ float  sBg[32];

    int tid = threadIdx.x;
    int t  = blockIdx.y;
    int m0 = blockIdx.x * 128;
    for (int i = tid; i < 1024; i += 256) sWg[i] = Wg[i];
    if (tid < 32) sBg[tid] = bg[tid];

    int lane = tid & 31, w = tid >> 5;
    bool hi = lane >= 16;
    int l16 = lane & 15;
    int o1 = hi ? 8 : 0;               // A-fragment split-K offset (ISA 7.12.2)
    int kb = hi ? 16 : 0;              // unused; B fragment also loads with o1

    const bf16_t* gA = AdjT + ((size_t)t*NE + m0)*NE;   // + n0
    const bf16_t* gY = yT + ((size_t)t*32)*NE;          // + n0

    v8f acc0 = {}; v8f acc1 = {};
    const int NSTEP = NE/32;

#if USE_TDM
    if (tid < 32) {   // prologue: prefetch tile 0 into buffer 0
        tdm_load_tile_2d((unsigned int)(uintptr_t)&sA[0][0][0], gA, NE, 32u, 128u);
        tdm_load_tile_2d((unsigned int)(uintptr_t)&sY[0][0][0], gY, NE, 32u, 32u);
    }
#endif

    for (int it = 0; it < NSTEP; it++) {
        int cur = it & 1;
#if USE_TDM
        if (tid < 32) {
            if (it + 1 < NSTEP) {      // issue next tile into the other buffer, then
                int n1 = (it + 1) * 32; // wait only for the CURRENT tile (in-order retire)
                tdm_load_tile_2d((unsigned int)(uintptr_t)&sA[cur^1][0][0], gA + n1, NE, 32u, 128u);
                tdm_load_tile_2d((unsigned int)(uintptr_t)&sY[cur^1][0][0], gY + n1, NE, 32u, 32u);
                __builtin_amdgcn_s_wait_tensorcnt(2);
            } else {
                __builtin_amdgcn_s_wait_tensorcnt(0);
            }
        }
#else
        {   // fallback: manual cooperative staging of tile `it` into buffer `cur`
            int n0 = it * 32;
            const unsigned int* pA = (const unsigned int*)(gA + n0);
            unsigned int* dA = (unsigned int*)&sA[cur][0][0];     // pitch 20 dwords
            #pragma unroll
            for (int q = 0; q < 8; q++) {
                int u = tid*8 + q;
                int r = u >> 4, cu = u & 15;
                dA[r*20 + cu] = pA[(size_t)r*(NE/2) + cu];
            }
            const unsigned int* pY = (const unsigned int*)(gY + n0);
            unsigned int* dY = (unsigned int*)&sY[cur][0][0];
            #pragma unroll
            for (int q = 0; q < 2; q++) {
                int u = tid*2 + q;
                int r = u >> 4, cu = u & 15;
                dY[r*20 + cu] = pY[(size_t)r*(NE/2) + cu];
            }
        }
#endif
        __syncthreads();   // tile `cur` visible to all waves

        const bf16_t* arow  = &sA[cur][(w << 4) + l16][0];
        const bf16_t* brow0 = &sY[cur][l16][0];
        const bf16_t* brow1 = &sY[cur][16 + l16][0];
        union { v16bf v; uint4 q[2]; } ua, ub0, ub1;
        ua.q[0]  = *(const uint4*)(arow + o1);
        ua.q[1]  = *(const uint4*)(arow + 16 + o1);
        ub0.q[0] = *(const uint4*)(brow0 + o1);
        ub0.q[1] = *(const uint4*)(brow0 + 16 + o1);
        ub1.q[0] = *(const uint4*)(brow1 + o1);
        ub1.q[1] = *(const uint4*)(brow1 + 16 + o1);

        acc0 = __builtin_amdgcn_wmma_f32_16x16x32_bf16(false, ua.v, false, ub0.v, (short)0, acc0, false, false);
        acc1 = __builtin_amdgcn_wmma_f32_16x16x32_bf16(false, ua.v, false, ub1.v, (short)0, acc1, false, false);

        __syncthreads();   // all reads of buffer `cur` done before it is refilled (it+2)
    }

    // D layout: VGPR v -> row M = v + 8*hi, col N = lane&15
    int mrow = (w << 4) + (hi ? 8 : 0);
    #pragma unroll
    for (int v = 0; v < 8; v++) {
        sMix[mrow + v][l16]      = acc0[v];
        sMix[mrow + v][16 + l16] = acc1[v];
    }
    __syncthreads();

    #pragma unroll
    for (int q = 0; q < 16; q++) {
        int idx = tid*16 + q;
        int m = idx >> 5, gg = idx & 31;
        float acc = sBg[gg];
        #pragma unroll
        for (int f = 0; f < 32; f++) acc += sMix[m][f]*sWg[f*32+gg];
        xout[((size_t)t*NE + m0 + m)*32 + gg] = fmaxf(acc, 0.f);
    }
}

// ---------------------------------------------------------------- final gather + MLP head
__global__ void __launch_bounds__(256) head_kernel(
    const int* __restrict__ info, const int* __restrict__ srv_attr, const int* __restrict__ svc_attr,
    const float* __restrict__ traF, const float* __restrict__ xpF, const float* __restrict__ xaF,
    const float* __restrict__ srv_tab0, const float* __restrict__ srv_lvl,
    const float* __restrict__ svc_tab0, const float* __restrict__ svc_lvl,
    const float* __restrict__ qW0, const float* __restrict__ qb0,
    const float* __restrict__ qW1, const float* __restrict__ qb1,
    const float* __restrict__ qW2, const float* __restrict__ qb2,
    const float* __restrict__ qW3, const float* __restrict__ qb3,
    const float* __restrict__ qW4, const float* __restrict__ qb4,
    const float* __restrict__ qW5, const float* __restrict__ qb5,
    float* __restrict__ out)
{
    __shared__ float sW0[64*112]; __shared__ float sB0[64];
    __shared__ float sW1[32*64];  __shared__ float sB1[32];
    __shared__ float sW2[16*32];  __shared__ float sB2[16];
    __shared__ float sW3[8*16];   __shared__ float sB3[8];
    __shared__ float sW4[4*8];    __shared__ float sB4[4];
    __shared__ float sW5[4];      __shared__ float sB5[1];
    int tid = threadIdx.x;
    for (int i = tid; i < 7168; i += 256) sW0[i] = qW0[i];
    for (int i = tid; i < 2048; i += 256) sW1[i] = qW1[i];
    for (int i = tid; i < 512;  i += 256) sW2[i] = qW2[i];
    if (tid < 128) sW3[tid] = qW3[tid];
    if (tid < 64)  sB0[tid] = qb0[tid];
    if (tid < 32) { sB1[tid] = qb1[tid]; sW4[tid] = qW4[tid]; }
    if (tid < 16)  sB2[tid] = qb2[tid];
    if (tid < 8)   sB3[tid] = qb3[tid];
    if (tid < 4) { sB4[tid] = qb4[tid]; sW5[tid] = qW5[tid]; }
    if (tid == 0)  sB5[0] = qb5[0];
    __syncthreads();

    int b = blockIdx.x*256 + tid;
    const int* ip = info + (size_t)b*4;
    int uid = ip[0], eid = ip[1], sid = ip[2];

    float a1[64];
    #pragma unroll
    for (int r = 0; r < 64; r++) a1[r] = sB0[r];
    {
        int c = 0;
        const float* tf = traF + (size_t)uid*16;
        for (int j = 0; j < 16; j++, c++) {
            float hv = tf[j];
            #pragma unroll
            for (int r = 0; r < 64; r++) a1[r] += sW0[r*112 + c]*hv;
        }
        size_t offE = ((size_t)15*NE + eid)*32;
        for (int j = 0; j < 32; j++, c++) {
            float hv = 0.5f*(xpF[offE + j] + xaF[offE + j]);
            #pragma unroll
            for (int r = 0; r < 64; r++) a1[r] += sW0[r*112 + c]*hv;
        }
        const int* sa = srv_attr + (size_t)eid*4;
        for (int j = 0; j < 8; j++, c++) {
            float hv = srv_tab0[(size_t)sa[0]*8 + j];
            #pragma unroll
            for (int r = 0; r < 64; r++) a1[r] += sW0[r*112 + c]*hv;
        }
        for (int k = 0; k < 3; k++)
            for (int j = 0; j < 8; j++, c++) {
                float hv = srv_lvl[k*80 + sa[1+k]*8 + j];
                #pragma unroll
                for (int r = 0; r < 64; r++) a1[r] += sW0[r*112 + c]*hv;
            }
        const int* va = svc_attr + (size_t)sid*4;
        for (int j = 0; j < 8; j++, c++) {
            float hv = svc_tab0[(size_t)va[0]*8 + j];
            #pragma unroll
            for (int r = 0; r < 64; r++) a1[r] += sW0[r*112 + c]*hv;
        }
        for (int k = 0; k < 3; k++)
            for (int j = 0; j < 8; j++, c++) {
                float hv = svc_lvl[k*80 + va[1+k]*8 + j];
                #pragma unroll
                for (int r = 0; r < 64; r++) a1[r] += sW0[r*112 + c]*hv;
            }
    }
    #pragma unroll
    for (int r = 0; r < 64; r++) a1[r] = fmaxf(a1[r], 0.f);

    float a2[32];
    #pragma unroll
    for (int r = 0; r < 32; r++) {
        float acc = sB1[r];
        #pragma unroll
        for (int cc = 0; cc < 64; cc++) acc += sW1[r*64+cc]*a1[cc];
        a2[r] = fmaxf(acc, 0.f);
    }
    float a3[16];
    #pragma unroll
    for (int r = 0; r < 16; r++) {
        float acc = sB2[r];
        #pragma unroll
        for (int cc = 0; cc < 32; cc++) acc += sW2[r*32+cc]*a2[cc];
        a3[r] = fmaxf(acc, 0.f);
    }
    float a4[8];
    #pragma unroll
    for (int r = 0; r < 8; r++) {
        float acc = sB3[r];
        #pragma unroll
        for (int cc = 0; cc < 16; cc++) acc += sW3[r*16+cc]*a3[cc];
        a4[r] = fmaxf(acc, 0.f);
    }
    float a5[4];
    #pragma unroll
    for (int r = 0; r < 4; r++) {
        float acc = sB4[r];
        #pragma unroll
        for (int cc = 0; cc < 8; cc++) acc += sW4[r*8+cc]*a4[cc];
        a5[r] = fmaxf(acc, 0.f);
    }
    float acc = sB5[0];
    #pragma unroll
    for (int cc = 0; cc < 4; cc++) acc += sW5[cc]*a5[cc];
    out[b] = acc;
}

// ---------------------------------------------------------------- launcher
extern "C" void kernel_launch(void* const* d_in, const int* in_sizes, int n_in,
                              void* d_out, int out_size, void* d_ws, size_t ws_size,
                              hipStream_t stream) {
    const float* edge     = (const float*)d_in[0];
    const float* loadx    = (const float*)d_in[1];
    const float* svcx     = (const float*)d_in[2];
    const float* tra      = (const float*)d_in[3];
    const int*   info     = (const int*)  d_in[4];
    const int*   srv_attr = (const int*)  d_in[5];
    const int*   svc_attr = (const int*)  d_in[6];
    const float* usr_tab  = (const float*)d_in[7];
    const float* srv_tab0 = (const float*)d_in[8];
    const float* srv_lvl  = (const float*)d_in[9];
    const float* svc_tab0 = (const float*)d_in[10];
    const float* svc_lvl  = (const float*)d_in[11];
    const float* lstm_Wih = (const float*)d_in[12];
    const float* lstm_Whh = (const float*)d_in[13];
    const float* lstm_bih = (const float*)d_in[14];
    const float* lstm_bhh = (const float*)d_in[15];
    const float* E1_p = (const float*)d_in[16];
    const float* E2_p = (const float*)d_in[17];
    const float* pe_p_W = (const float*)d_in[18];
    const float* pe_p_b = (const float*)d_in[19];
    const float* E1_a = (const float*)d_in[20];
    const float* E2_a = (const float*)d_in[21];
    const float* pe_a_W = (const float*)d_in[22];
    const float* pe_a_b = (const float*)d_in[23];
    const float* proj_p_W = (const float*)d_in[24];
    const float* proj_p_b = (const float*)d_in[25];
    const float* proj_a_W = (const float*)d_in[26];
    const float* proj_a_b = (const float*)d_in[27];
    const float* Wt_p = (const float*)d_in[28];
    const float* bt_p = (const float*)d_in[29];
    const float* Wg_p = (const float*)d_in[30];
    const float* bg_p = (const float*)d_in[31];
    const float* Wt_a = (const float*)d_in[32];
    const float* bt_a = (const float*)d_in[33];
    const float* Wg_a = (const float*)d_in[34];
    const float* bg_a = (const float*)d_in[35];
    const float* qW[6]; const float* qb[6];
    for (int i = 0; i < 6; i++) { qW[i] = (const float*)d_in[36 + 2*i]; qb[i] = (const float*)d_in[37 + 2*i]; }

    char* ws = (char*)d_ws;
    size_t off = 0;
    auto carve = [&](size_t bytes) -> void* {
        off = (off + 255) & ~(size_t)255;
        void* p = ws + off; off += bytes; return p;
    };
    float*  traF  = (float*)carve((size_t)NU*16*4);
    float*  epP   = (float*)carve((size_t)K_T*NE*8*4);
    float*  epA   = (float*)carve((size_t)K_T*NE*8*4);
    float*  Tp    = (float*)carve((size_t)K_T*NE*8*4);
    float*  Ta    = (float*)carve((size_t)K_T*NE*8*4);
    float*  xpA   = (float*)carve((size_t)K_T*NE*32*4);
    float*  xpB   = (float*)carve((size_t)K_T*NE*32*4);
    float*  xaA   = (float*)carve((size_t)K_T*NE*32*4);
    float*  xaB   = (float*)carve((size_t)K_T*NE*32*4);
    bf16_t* yT    = (bf16_t*)carve((size_t)K_T*32*NE*2);
    bf16_t* AdjW  = (bf16_t*)carve((size_t)K_T*NE*NE*2);   // row-major scratch
    bf16_t* AdjTp = (bf16_t*)carve((size_t)K_T*NE*NE*2);
    bf16_t* AdjTa = (bf16_t*)carve((size_t)K_T*NE*NE*2);

    lstm_kernel<<<NU/256, 256, 0, stream>>>(tra, usr_tab, lstm_Wih, lstm_Whh, lstm_bih, lstm_bhh, traF);
    ep_kernel<<<(K_T*NE)/256, 256, 0, stream>>>(edge, pe_p_W, pe_p_b, pe_a_W, pe_a_b, epP, epA);
    tmat_kernel<<<dim3(NE/256, K_T), 256, 0, stream>>>(epP, E1_p, E2_p, Tp);
    tmat_kernel<<<dim3(NE/256, K_T), 256, 0, stream>>>(epA, E1_a, E2_a, Ta);
    proj_kernel<<<(K_T*NE)/256, 256, 0, stream>>>(loadx, svcx, proj_p_W, proj_p_b, proj_a_W, proj_a_b, xpA, xaA);

    adj_kernel<<<dim3(NE, K_T), 256, 0, stream>>>(Tp, epP, AdjW);
    transpose_kernel<<<dim3(NE/64, NE/64, K_T), 256, 0, stream>>>(AdjW, AdjTp);
    adj_kernel<<<dim3(NE, K_T), 256, 0, stream>>>(Ta, epA, AdjW);
    transpose_kernel<<<dim3(NE/64, NE/64, K_T), 256, 0, stream>>>(AdjW, AdjTa);

    const int dil[4] = {1, 2, 1, 2};

    float* xin = xpA; float* xout = xpB;
    for (int i = 0; i < 4; i++) {
        conv_kernel<<<(K_T*NE)/256, 256, 0, stream>>>(xin, Wt_p + (size_t)i*3*32*32, bt_p + i*32, dil[i], yT);
        mix_kernel<<<dim3(NE/128, K_T), 256, 0, stream>>>(AdjTp, yT, Wg_p + (size_t)i*32*32, bg_p + i*32, xout);
        float* t = xin; xin = xout; xout = t;
    }
    float* xpFinal = xin;

    xin = xaA; xout = xaB;
    for (int i = 0; i < 4; i++) {
        conv_kernel<<<(K_T*NE)/256, 256, 0, stream>>>(xin, Wt_a + (size_t)i*3*32*32, bt_a + i*32, dil[i], yT);
        mix_kernel<<<dim3(NE/128, K_T), 256, 0, stream>>>(AdjTa, yT, Wg_a + (size_t)i*32*32, bg_a + i*32, xout);
        float* t = xin; xin = xout; xout = t;
    }
    float* xaFinal = xin;

    head_kernel<<<BATCH/256, 256, 0, stream>>>(
        info, srv_attr, svc_attr, traF, xpFinal, xaFinal,
        srv_tab0, srv_lvl, svc_tab0, svc_lvl,
        qW[0], qb[0], qW[1], qb[1], qW[2], qb[2], qW[3], qb[3], qW[4], qb[4], qW[5], qb[5],
        (float*)d_out);
}
